// LSTM_53695681134710
// MI455X (gfx1250) — matrix operand order repaired and checked by reading, not compiled, not run on hardware
//
#include <hip/hip_runtime.h>

// Problem constants (match reference)
#define LSTM_B   64
#define LSTM_T   512
#define LSTM_IN  512
#define LSTM_H   1024
#define LSTM_K   (LSTM_IN + LSTM_H)   // 1536 = fused [x_t ; h] K dimension
#define LSTM_O   128
#define NBLK     64                   // persistent blocks (one 16-col tile of H each)
#define KCHUNK   512                  // K split: 3 chunks of 512

typedef __attribute__((ext_vector_type(16))) __bf16 v16bf;
typedef __attribute__((ext_vector_type(8)))  __bf16 v8bf;
typedef __attribute__((ext_vector_type(8)))  float  v8f;

#define SHUF16(lo, hi) \
  __builtin_shufflevector(lo, hi, 0,1,2,3,4,5,6,7,8,9,10,11,12,13,14,15)

__device__ __forceinline__ unsigned short f2bf(float f) {
  unsigned u = __builtin_bit_cast(unsigned, f);
  u += 0x7fffu + ((u >> 16) & 1u);          // round-to-nearest-even
  return (unsigned short)(u >> 16);
}
__device__ __forceinline__ float bf2f(unsigned short h) {
  return __builtin_bit_cast(float, ((unsigned)h) << 16);
}
__device__ __forceinline__ float sigm(float x) {
  return 1.0f / (1.0f + __expf(-x));
}

// ---------------- Phase 0: packing / init ----------------

// Wcat[r, k] (bf16): k<512 -> W_ih[r,k], else W_hh[r,k-512]
__global__ void k_pack_weights(const float* __restrict__ W_ih,
                               const float* __restrict__ W_hh,
                               unsigned short* __restrict__ Wcat) {
  const size_t n = (size_t)4 * LSTM_H * LSTM_K;
  for (size_t i = (size_t)blockIdx.x * blockDim.x + threadIdx.x; i < n;
       i += (size_t)gridDim.x * blockDim.x) {
    size_t r = i / LSTM_K, k = i % LSTM_K;
    float v = (k < LSTM_IN) ? W_ih[r * LSTM_IN + k]
                            : W_hh[r * LSTM_H + (k - LSTM_IN)];
    Wcat[i] = f2bf(v);
  }
}

__global__ void k_pack_x(const float* __restrict__ x,
                         unsigned short* __restrict__ Xbf) {
  const size_t n = (size_t)LSTM_B * LSTM_T * LSTM_IN;
  for (size_t i = (size_t)blockIdx.x * blockDim.x + threadIdx.x; i < n;
       i += (size_t)gridDim.x * blockDim.x)
    Xbf[i] = f2bf(x[i]);
}

__global__ void k_bias_init(const float* __restrict__ b_ih,
                            const float* __restrict__ b_hh,
                            float* __restrict__ bias,
                            unsigned short* __restrict__ H0,
                            float* __restrict__ Cst,
                            unsigned* __restrict__ barrier_ctr) {
  int i = blockIdx.x * blockDim.x + threadIdx.x;
  if (i == 0) *barrier_ctr = 0u;
  if (i < 4 * LSTM_H) bias[i] = b_ih[i] + b_hh[i];
  if (i < LSTM_B * LSTM_H) { H0[i] = 0; Cst[i] = 0.0f; }
}

// ---------------- device-wide barrier (monotonic counter, 64 co-resident blocks) --
__device__ __forceinline__ void grid_sync(unsigned* ctr, unsigned target) {
  __syncthreads();
  if (threadIdx.x == 0) {
    __hip_atomic_fetch_add(ctr, 1u, __ATOMIC_RELEASE, __HIP_MEMORY_SCOPE_AGENT);
    while (__hip_atomic_load(ctr, __ATOMIC_ACQUIRE, __HIP_MEMORY_SCOPE_AGENT) < target)
      __builtin_amdgcn_s_sleep(2);
  }
  __syncthreads();
  __builtin_amdgcn_fence(__ATOMIC_ACQUIRE, "agent");  // invalidate stale L0 lines
}

// ---------------- Phase 1: persistent fused LSTM over all 512 steps ----------------
// 64 blocks x 384 threads (12 waves). Block owns hidden cols [16*bx, 16*bx+16).
// Wave = (kchunk, mwave): kchunk 0 -> A = x_t, kchunk 1/2 -> A = h halves;
// mwave selects 16 batch rows. Each wave computes 4 gate tiles over its 512-K
// chunk (16 chained WMMAs); partials reduced in LDS; kchunk-0 waves apply the
// LSTM cell update. One device-wide barrier per timestep; h ping-pongs.
__global__ void __launch_bounds__(384)
k_lstm_persistent(const unsigned short* __restrict__ Xbf,   // [B,T,512] bf16
                  const unsigned short* __restrict__ Wcat,  // [4H,1536] bf16
                  const float* __restrict__ bias,           // [4H]
                  unsigned short* __restrict__ Ha,          // h ping [B,H] bf16
                  unsigned short* __restrict__ Hb,          // h pong [B,H] bf16
                  float* __restrict__ Cst,                  // [B,H] f32 (in-place)
                  unsigned* __restrict__ barrier_ctr) {
  __shared__ v8f red[8 * 4 * 32];   // 32 KB: 8 partial slots x 4 gates x 32 lanes

  const int lane   = threadIdx.x & 31;
  const int wave   = threadIdx.x >> 5;
  const int mwave  = wave & 3;
  const int kchunk = wave >> 2;        // 0: x, 1: h[0:512), 2: h[512:1024)
  const int hi     = lane >> 4;
  const int l15    = lane & 15;
  const int M0     = mwave * 16;
  const int N0     = blockIdx.x * 16;
  const int n      = N0 + l15;
  const int arow   = M0 + l15;

  // Weight column pointers for this wave's K chunk (B 32x16 in this code's contiguous K order, not ISA table order:
  // lane -> col N=l15, 16 contiguous K starting at hi*16). Wcat row = B column.
  const size_t wko = (size_t)kchunk * KCHUNK + (size_t)hi * 16;
  const __bf16* b0 = (const __bf16*)Wcat + ((size_t)(0 * LSTM_H + n)) * LSTM_K + wko;
  const __bf16* b1 = (const __bf16*)Wcat + ((size_t)(1 * LSTM_H + n)) * LSTM_K + wko;
  const __bf16* b2 = (const __bf16*)Wcat + ((size_t)(2 * LSTM_H + n)) * LSTM_K + wko;
  const __bf16* b3 = (const __bf16*)Wcat + ((size_t)(3 * LSTM_H + n)) * LSTM_K + wko;

  const float bi  = bias[n];
  const float bf_ = bias[LSTM_H + n];
  const float bg  = bias[2 * LSTM_H + n];
  const float bo  = bias[3 * LSTM_H + n];

  const __bf16* ax_base = (const __bf16*)Xbf + (size_t)arow * LSTM_T * LSTM_IN;

  unsigned short* Hin  = Ha;
  unsigned short* Hout = Hb;
  unsigned target = NBLK;

#pragma unroll 1
  for (int t = 0; t < LSTM_T; ++t) {
    v8f acc0 = {}, acc1 = {}, acc2 = {}, acc3 = {};

    // A row pointer (A 16x32 layout: lane -> row M=l15, 16 contiguous K
    // starting at hi*16, matching B)
    const __bf16* ap = (kchunk == 0)
        ? (ax_base + (size_t)t * LSTM_IN)
        : ((const __bf16*)Hin + (size_t)arow * LSTM_H + (size_t)(kchunk - 1) * KCHUNK);

#pragma unroll 4
    for (int kt = 0; kt < KCHUNK / 32; ++kt) {
      const int K0 = kt * 32;
      v16bf a  = SHUF16(*(const v8bf*)(ap + K0 + hi * 16),
                        *(const v8bf*)(ap + K0 + hi * 16 + 8));
      v16bf w0 = SHUF16(*(const v8bf*)(b0 + K0), *(const v8bf*)(b0 + K0 + 8));
      v16bf w1 = SHUF16(*(const v8bf*)(b1 + K0), *(const v8bf*)(b1 + K0 + 8));
      v16bf w2 = SHUF16(*(const v8bf*)(b2 + K0), *(const v8bf*)(b2 + K0 + 8));
      v16bf w3 = SHUF16(*(const v8bf*)(b3 + K0), *(const v8bf*)(b3 + K0 + 8));
      acc0 = __builtin_amdgcn_wmma_f32_16x16x32_bf16(false, a, false, w0, (short)0, acc0, false, false);
      acc1 = __builtin_amdgcn_wmma_f32_16x16x32_bf16(false, a, false, w1, (short)0, acc1, false, false);
      acc2 = __builtin_amdgcn_wmma_f32_16x16x32_bf16(false, a, false, w2, (short)0, acc2, false, false);
      acc3 = __builtin_amdgcn_wmma_f32_16x16x32_bf16(false, a, false, w3, (short)0, acc3, false, false);
    }

    // Prefetch next timestep's x fragment (global_prefetch_b8) while we wait.
    if (kchunk == 0 && t + 1 < LSTM_T)
      __builtin_prefetch((const void*)(ax_base + (size_t)(t + 1) * LSTM_IN + hi * 8), 0, 0);

    // K-split reduction through LDS (f32, no precision loss)
    if (kchunk != 0) {
      const int slot = (kchunk - 1) * 4 + mwave;
      red[(slot * 4 + 0) * 32 + lane] = acc0;
      red[(slot * 4 + 1) * 32 + lane] = acc1;
      red[(slot * 4 + 2) * 32 + lane] = acc2;
      red[(slot * 4 + 3) * 32 + lane] = acc3;
    }
    __syncthreads();

    if (kchunk == 0) {
#pragma unroll
      for (int p = 0; p < 2; ++p) {
        const int slot = p * 4 + mwave;
        acc0 += red[(slot * 4 + 0) * 32 + lane];
        acc1 += red[(slot * 4 + 1) * 32 + lane];
        acc2 += red[(slot * 4 + 2) * 32 + lane];
        acc3 += red[(slot * 4 + 3) * 32 + lane];
      }
      // fused LSTM cell update (C/D layout: lane -> col n=l15, vgpr v -> row hi*8+v)
#pragma unroll
      for (int v = 0; v < 8; ++v) {
        const int m = M0 + hi * 8 + v;
        const size_t idx = (size_t)m * LSTM_H + n;
        const float iv = sigm(acc0[v] + bi);
        const float fv = sigm(acc1[v] + bf_);
        const float gv = tanhf(acc2[v] + bg);
        const float ov = sigm(acc3[v] + bo);
        const float c = fv * Cst[idx] + iv * gv;
        Cst[idx] = c;
        Hout[idx] = f2bf(ov * tanhf(c));
      }
    }

    // Device-wide barrier: all blocks finished reading Hin & writing Hout.
    grid_sync(barrier_ctr, target);
    target += NBLK;

    unsigned short* tmp = Hout; Hout = Hin; Hin = tmp;
  }
}

// ---------------- Phase 2: classifier (tiny: 64x128x1024, f32) ----------------
__global__ void __launch_bounds__(256)
k_classifier(const unsigned short* __restrict__ Hbuf,  // [B,H] bf16
             const float* __restrict__ Wc,             // [O,H] f32
             const float* __restrict__ bc,             // [O]
             float* __restrict__ out) {                // [B,O]
  int j = blockIdx.x * blockDim.x + threadIdx.x;
  if (j >= LSTM_B * LSTM_O) return;
  int b = j >> 7;
  int o = j & (LSTM_O - 1);
  const unsigned short* h = Hbuf + (size_t)b * LSTM_H;
  const float* w = Wc + (size_t)o * LSTM_H;
  float s = bc[o];
#pragma unroll 8
  for (int k = 0; k < LSTM_H; ++k) s = fmaf(bf2f(h[k]), w[k], s);
  out[j] = s;
}

// ---------------- Host launcher ----------------
extern "C" void kernel_launch(void* const* d_in, const int* in_sizes, int n_in,
                              void* d_out, int out_size, void* d_ws, size_t ws_size,
                              hipStream_t stream) {
  (void)in_sizes; (void)n_in; (void)out_size; (void)ws_size;
  const float* x     = (const float*)d_in[0];
  const float* W_ih  = (const float*)d_in[1];
  const float* W_hh  = (const float*)d_in[2];
  const float* b_ih  = (const float*)d_in[3];
  const float* b_hh  = (const float*)d_in[4];
  const float* W_clf = (const float*)d_in[5];
  const float* b_clf = (const float*)d_in[6];
  float* out = (float*)d_out;

  // Workspace carve-out (~47 MB total)
  size_t off = 0;
  auto alloc = [&](size_t bytes) -> void* {
    void* p = (char*)d_ws + off;
    off += (bytes + 255) & ~(size_t)255;
    return p;
  };
  unsigned short* Wcat  = (unsigned short*)alloc((size_t)4 * LSTM_H * LSTM_K * 2);
  unsigned short* Xbf   = (unsigned short*)alloc((size_t)LSTM_B * LSTM_T * LSTM_IN * 2);
  float*          bias  = (float*)alloc((size_t)4 * LSTM_H * 4);
  unsigned short* Hping = (unsigned short*)alloc((size_t)LSTM_B * LSTM_H * 2);
  unsigned short* Hpong = (unsigned short*)alloc((size_t)LSTM_B * LSTM_H * 2);
  float*          Cst   = (float*)alloc((size_t)LSTM_B * LSTM_H * 4);
  unsigned*       bctr  = (unsigned*)alloc(sizeof(unsigned));

  // Phase 0: pack weights/x to bf16, fold biases, zero h0/c0/barrier
  k_pack_weights<<<4096, 256, 0, stream>>>(W_ih, W_hh, Wcat);
  k_pack_x<<<8192, 256, 0, stream>>>(x, Xbf);
  k_bias_init<<<(LSTM_B * LSTM_H + 255) / 256, 256, 0, stream>>>(
      b_ih, b_hh, bias, Hping, Cst, bctr);

  // Phase 1: one persistent kernel runs all 512 timesteps.
  // t even writes Hpong, t odd writes Hping -> after 512 steps final h is Hping.
  k_lstm_persistent<<<NBLK, 384, 0, stream>>>(Xbf, Wcat, bias, Hping, Hpong, Cst, bctr);

  // Phase 2: classifier on final h (in Hping after an even number of steps)
  k_classifier<<<(LSTM_B * LSTM_O + 255) / 256, 256, 0, stream>>>(
      Hping, W_clf, b_clf, out);
}
